// MHScaledDotProductSoftmaxAttention_455266534013
// MI455X (gfx1250) — matrix operand order repaired and checked by reading, not compiled, not run on hardware
//
#include <hip/hip_runtime.h>

// ---------------------------------------------------------------------------
// MI455X (gfx1250) multi-head self-Gram attention.
//   out = concat(X, S @ softmax_col(relu(S^T S / 8 + bias))^T) per (n, head).
// One workgroup per (sample, head): full 64x1024 slice resident in LDS as
// bf16 (S row-major + S^T), all GEMMs via v_wmma_f32_16x16x32_bf16.
// ---------------------------------------------------------------------------

typedef __bf16 bf16_t;
typedef __attribute__((ext_vector_type(16))) __bf16 v16bf;
typedef __attribute__((ext_vector_type(8)))  __bf16 v8bf;
typedef __attribute__((ext_vector_type(8)))  float  v8f;

#define NPOP   1024
#define NWIN   1024
#define PDIM   64          // pop per head
#define ST_STRIDE 72       // S^T row stride in halves (64 + 8 pad, 16B-aligned rows)
#define S_STRIDE  1032     // S row stride in halves (1024 + 8 pad, 16B-aligned rows)

union V16U { v16bf v; v8bf h[2]; };

// WMMA A-operand (16x32 bf16, rows M, K contiguous in memory).
// Layout (ISA 7.12.2): lanes 0-15 -> M=lane, K = {kc..kc+7, kc+16..kc+23};
//                      lanes 16-31 -> M=lane-16, K = {kc+8..kc+15, kc+24..kc+31}.
__device__ __forceinline__ v16bf wmma_ldA(const bf16_t* base, int row, int stride,
                                          int kc, int lane) {
  int m = lane & 15, hi = lane >> 4;
  const bf16_t* p = base + (row + m) * stride + kc + hi * 8;
  V16U u;
  u.h[0] = *(const v8bf*)p;          // K = kc+hi*8 .. +7
  u.h[1] = *(const v8bf*)(p + 16);   // K = kc+hi*8+16 .. +23
  return u.v;
}

// WMMA B-operand (32x16 bf16, element [k,n] at n*stride + k, K contiguous).
// Layout: lanes 0-15 -> N=lane, K = {kc..kc+7, kc+16..kc+23}; lanes 16-31 -> N=lane-16, K = {kc+8..kc+15, kc+24..kc+31}.
__device__ __forceinline__ v16bf wmma_ldB(const bf16_t* base, int col, int stride,
                                          int kc, int lane) {
  int n = lane & 15, hi = lane >> 4;
  const bf16_t* p = base + (col + n) * stride + kc + hi * 8;
  V16U u;
  u.h[0] = *(const v8bf*)p;
  u.h[1] = *(const v8bf*)(p + 16);
  return u.v;
}

#define WMMA_BF16(a, b, c) \
  __builtin_amdgcn_wmma_f32_16x16x32_bf16(false, (a), false, (b), (short)0, (c), false, false)

extern "C" __global__ __launch_bounds__(256, 1)
void mh_attn_kernel(const float* __restrict__ X, const float* __restrict__ bias,
                    float* __restrict__ out) {
  extern __shared__ char smem[];
  bf16_t* ST = (bf16_t*)smem;                       // [1024][ST_STRIDE]  S^T (144 KB)
  bf16_t* Sm = ST + NWIN * ST_STRIDE;               // [64][S_STRIDE]     S   (129 KB)
  float*  Z  = (float*)(Sm + PDIM * S_STRIDE);      // [1024]             (4 KB)
  bf16_t* Pt = (bf16_t*)(Z + NWIN);                 // [8 waves][16*32]   (8 KB)

  const int blk  = blockIdx.x;
  const int n    = blk >> 4;
  const int h    = blk & 15;
  const int tid  = threadIdx.x;
  const int lane = tid & 31;
  const int wave = tid >> 5;
  const float bias0 = bias[0];

  const float* Xg     = X   + ((size_t)n * NPOP + h * PDIM) * NWIN;
  float*       outTop = out + ((size_t)n * 2 * NPOP + h * PDIM) * NWIN;
  float*       outBot = out + ((size_t)n * 2 * NPOP + NPOP + h * PDIM) * NWIN;

  // ---------- Phase 1: load slice, emit X-copy, stage S and S^T as bf16 ----
  for (int i = tid; i < PDIM * NWIN; i += 256) {
    int p = i >> 10, w = i & 1023;                  // coalesced in w
    float v = Xg[(size_t)p * NWIN + w];
    outTop[(size_t)p * NWIN + w] = v;
    bf16_t b = (bf16_t)v;
    Sm[p * S_STRIDE + w] = b;
    ST[w * ST_STRIDE + p] = b;
  }
  __syncthreads();

  // ---------- Phase 2: Z[b] = sum_a exp(relu(Gram[a,b]/8 + bias)) ---------
  // Wave wv owns w-tiles {wv, wv+8, ..., wv+56} (disjoint columns).
  for (int wt = 0; wt < 8; ++wt) {
    const int wcol = (wave + wt * 8) * 16;
    v8f zacc = {};
    for (int li = 0; li < 64; ++li) {
      v8f g = {};
#pragma unroll
      for (int kc = 0; kc < 64; kc += 32) {
        v16bf a = wmma_ldA(ST, li * 16, ST_STRIDE, kc, lane);
        v16bf b = wmma_ldB(ST, wcol,    ST_STRIDE, kc, lane);
        g = WMMA_BF16(a, b, g);
      }
#pragma unroll
      for (int j = 0; j < 8; ++j)
        zacc[j] += __expf(fmaxf(fmaf(g[j], 0.125f, bias0), 0.0f));
    }
    float zp = ((zacc[0] + zacc[4]) + (zacc[1] + zacc[5])) +
               ((zacc[2] + zacc[6]) + (zacc[3] + zacc[7]));
    zp += __shfl_xor(zp, 16, 32);                   // lanes L and L+16 share a column
    if (lane < 16) Z[wcol + lane] = zp;
  }
  __syncthreads();
  for (int b = tid; b < NWIN; b += 256) Z[b] = 1.0f / Z[b];
  __syncthreads();

  // ---------- Phase 3: D[c,a] = sum_b S[c,b] * P[a,b],  P = E * Zinv[b] ----
  bf16_t* Ptw = Pt + wave * (16 * 32);              // per-wave B-operand staging
  const int cn  = lane & 15;
  const int chi = lane >> 4;
  for (int at = 0; at < 8; ++at) {
    const int an = (wave + at * 8) * 16;            // a-tile base (disjoint per wave)
    v8f d0 = {}, d1 = {}, d2 = {}, d3 = {};
    for (int bt = 0; bt < 64; ++bt) {
      const int bb = bt * 16;
      // recompute Gram tile [a-rows an.., b-cols bb..]
      v8f g = {};
#pragma unroll
      for (int kc = 0; kc < 64; kc += 32) {
        v16bf a = wmma_ldA(ST, an, ST_STRIDE, kc, lane);
        v16bf b = wmma_ldB(ST, bb, ST_STRIDE, kc, lane);
        g = WMMA_BF16(a, b, g);
      }
      const float zinv = Z[bb + cn];                // lane's column b = bb + cn
      const int hsel = (bt & 1) * 16;
#pragma unroll
      for (int j = 0; j < 8; ++j) {
        float pv = __expf(fmaxf(fmaf(g[j], 0.125f, bias0), 0.0f)) * zinv;
        Ptw[(j + 8 * chi) * 32 + hsel + cn] = (bf16_t)pv;  // Pt[a_local][b_local]
      }
      if (bt & 1) {                                 // have a full K=32 chunk of b
        const int chunk = (bt >> 1) * 32;
        v16bf bp = wmma_ldB(Ptw, 0, 32, 0, lane);   // P tile as B-operand
        v16bf a0 = wmma_ldA(Sm,  0, S_STRIDE, chunk, lane);
        v16bf a1 = wmma_ldA(Sm, 16, S_STRIDE, chunk, lane);
        v16bf a2 = wmma_ldA(Sm, 32, S_STRIDE, chunk, lane);
        v16bf a3 = wmma_ldA(Sm, 48, S_STRIDE, chunk, lane);
        d0 = WMMA_BF16(a0, bp, d0);
        d1 = WMMA_BF16(a1, bp, d1);
        d2 = WMMA_BF16(a2, bp, d2);
        d3 = WMMA_BF16(a3, bp, d3);
      }
    }
    // store D tiles: C layout -> lane column a = an+cn, rows m = j + 8*chi
    const int a = an + cn;
#pragma unroll
    for (int j = 0; j < 8; ++j) {
      const int m = j + 8 * chi;
      outBot[(size_t)(m)      * NWIN + a] = d0[j];
      outBot[(size_t)(16 + m) * NWIN + a] = d1[j];
      outBot[(size_t)(32 + m) * NWIN + a] = d2[j];
      outBot[(size_t)(48 + m) * NWIN + a] = d3[j];
    }
  }
}

extern "C" void kernel_launch(void* const* d_in, const int* in_sizes, int n_in,
                              void* d_out, int out_size, void* d_ws, size_t ws_size,
                              hipStream_t stream) {
  const float* X    = (const float*)d_in[0];
  const float* bias = (const float*)d_in[1];
  float* out        = (float*)d_out;

  // LDS: S^T (144 KB) + S (129 KB) + Z (4 KB) + per-wave P staging (8 KB)
  const size_t smem = (size_t)NWIN * ST_STRIDE * sizeof(bf16_t)
                    + (size_t)PDIM * S_STRIDE  * sizeof(bf16_t)
                    + (size_t)NWIN * sizeof(float)
                    + (size_t)8 * 16 * 32 * sizeof(bf16_t);   // = 291840 B < 320 KB

  (void)hipFuncSetAttribute((const void*)mh_attn_kernel,
                            hipFuncAttributeMaxDynamicSharedMemorySize, (int)smem);

  dim3 grid(256);   // (n, h) pairs: 16 samples x 16 heads
  dim3 block(256);  // 8 wave32
  mh_attn_kernel<<<grid, block, smem, stream>>>(X, bias, out);
}
